// CNN_binary_notlast_69260642615467
// MI455X (gfx1250) — 8-bit matrix operand order repaired and checked by reading, compiled, not run on hardware
//
#include <hip/hip_runtime.h>
#include <hip/hip_bf16.h>
#include <stdint.h>

typedef int  v8i __attribute__((ext_vector_type(8)));
typedef unsigned int u32;
typedef u32  v4u __attribute__((ext_vector_type(4)));
typedef signed char i8;

#define BN_EPS 1e-5f

#define OFF_A1   0u
#define OFF_A2   512u
#define OFF_A3   8704u
#define OFF_A4   33280u
#define OFF_T1   131584u
#define OFF_T2   131712u
#define OFF_T3   131968u
#define OFF_S4   132480u
#define OFF_D4   132992u
#define OFF_H1S  135168u
#define OFF_H2S  (OFF_H1S + 53477376u)
#define OFF_H3S  (OFF_H2S + 106954752u)

__device__ __forceinline__ i8 sgnf(float v){ return v>0.f ? (i8)1 : (v<0.f ? (i8)-1 : (i8)0); }

__device__ __forceinline__ int dot4ss(u32 a, u32 b, int c){
#if __has_builtin(__builtin_amdgcn_sudot4)
    return __builtin_amdgcn_sudot4(true, (int)a, true, (int)b, c, false);
#elif __has_builtin(__builtin_amdgcn_sdot4)
    return __builtin_amdgcn_sdot4((int)a, (int)b, c, false);
#else
    int r = c;
    r += (int)(i8)(a      ) * (int)(i8)(b      );
    r += (int)(i8)(a >>  8) * (int)(i8)(b >>  8);
    r += (int)(i8)(a >> 16) * (int)(i8)(b >> 16);
    r += (int)(i8)(a >> 24) * (int)(i8)(b >> 24);
    return r;
#endif
}

__global__ void k_prep_params(
    const float* b1,const float* g1,const float* be1,const float* m1,const float* v1,
    const float* b2,const float* g2,const float* be2,const float* m2,const float* v2,
    const float* b3,const float* g3,const float* be3,const float* m3,const float* v3,
    const float* b4,const float* g4,const float* be4,const float* m4,const float* v4,
    int* T1, int* T2, int* T3, float* S4, float* D4)
{
    int c = threadIdx.x;
    if (c < 32) {
        float t = m1[c] - be1[c]*sqrtf(v1[c]+BN_EPS)/g1[c] - b1[c];
        T1[c] = (int)floorf(t) + 1;
    }
    if (c < 64) {
        float t = m2[c] - be2[c]*sqrtf(v2[c]+BN_EPS)/g2[c] - b2[c];
        T2[c] = (int)floorf(t) + 1;
    }
    if (c < 128) {
        float t = m3[c] - be3[c]*sqrtf(v3[c]+BN_EPS)/g3[c] - b3[c];
        T3[c] = (int)floorf(t) + 1;
        float sc = g4[c]/sqrtf(v4[c]+BN_EPS);
        S4[c] = sc;
        D4[c] = (b4[c]-m4[c])*sc + be4[c];
    }
}

__global__ void k_prep_weights(const float* w1,const float* w2,const float* w3,const float* w4,
                               i8* A1, i8* A2, i8* A3, i8* A4)
{
    int i = blockIdx.x*256 + threadIdx.x;
    if (i < 384) {
        int oc = i/12, t = i%12;
        A1[i] = (t < 9) ? sgnf(w1[oc*9 + t]) : (i8)0;
    }
    if (i < 8192) {
        int oc = i >> 7, k = i & 127;
        i8 v = 0;
        if (k < 64)      { int kw = k>>5, cin = k&31; v = sgnf(w2[oc*96 + cin*3 + kw]); }
        else if (k < 96) { int cin = k-64;            v = sgnf(w2[oc*96 + cin*3 + 2 ]); }
        A2[i] = v;
    }
    if (i < 24576) {
        int oc = i/192, k = i%192;
        int kw = k>>6, cin = k&63;
        A3[i] = sgnf(w3[oc*192 + cin*3 + kw]);
    }
    if (i < 98304) {
        int oc = i/768, k = i%768;
        int h = k>>7, cin = k&127;
        A4[i] = sgnf(w4[(oc*128 + cin)*6 + h]);
    }
}

__global__ void k_layer1(const float* __restrict__ x, const i8* __restrict__ A1,
                         const int* __restrict__ T1, i8* __restrict__ h1s)
{
    __shared__ __align__(16) i8 srow[8][144];
    __shared__ u32 wl[96];
    __shared__ int Tl[32];
    int tid = threadIdx.x;
    for (int i = tid; i < 96; i += 256) wl[i] = ((const u32*)A1)[i];
    if (tid < 32) Tl[tid] = T1[tid];

    int row_l = tid >> 5, lane = tid & 31;
    int row   = blockIdx.x*8 + row_l;
    const float* xr = x + row*128;
#pragma unroll
    for (int q = 0; q < 4; q++) {
        float xv = xr[lane*4 + q];
        srow[row_l][4 + lane*4 + q] = sgnf(xv);
    }
    if (lane < 4)               srow[row_l][lane]       = 0;
    if (lane >= 4 && lane < 16) srow[row_l][128 + lane] = 0;
    __syncthreads();

    u32 w0 = *(const u32*)&srow[row_l][4*lane + 0];
    u32 w1 = *(const u32*)&srow[row_l][4*lane + 4];
    u32 w2 = *(const u32*)&srow[row_l][4*lane + 8];
    u32 va = (w0 >> 16) | (w1 << 16);
    u32 vb = (w1 >> 16) | (w2 << 16);
    int b8  = (int)(i8)(w2 & 0xff);
    int b10 = (int)(i8)((w2 >> 16) & 0xff);

    i8* out = h1s + (size_t)row*1088;
    u32* o32 = (u32*)(out + (1 + lane)*32);
#pragma unroll
    for (int ocq = 0; ocq < 8; ocq++) {
        u32 p = 0;
#pragma unroll
        for (int b = 0; b < 4; b++) {
            int oc = ocq*4 + b;
            u32 u0 = wl[oc*3 + 0], u1 = wl[oc*3 + 1];
            int ws8 = (int)(i8)(wl[oc*3 + 2] & 0xff);
            int a0 = dot4ss(w1, u1, dot4ss(w0, u0, b8 * ws8));
            int a1 = dot4ss(vb, u1, dot4ss(va, u0, b10 * ws8));
            int am = a0 > a1 ? a0 : a1;
            i8 s = (am >= Tl[oc]) ? (i8)1 : (i8)-1;
            p |= ((u32)(unsigned char)s) << (8*b);
        }
        o32[ocq] = p;
    }
    if (tid < 128) {
        int r = tid >> 4, rem = tid & 15;
        int wp = (rem >> 3) ? 33 : 0, q = rem & 7;
        ((u32*)(h1s + (size_t)(blockIdx.x*8 + r)*1088 + wp*32))[q] = 0u;
    }
}

__global__ void k_layer2(const i8* __restrict__ h1s, const i8* __restrict__ A2,
                         const int* __restrict__ T2, i8* __restrict__ h2s)
{
    __shared__ __align__(16) i8 a2l[64*128];
    __shared__ __align__(16) i8 slab[4*34*32];
    __shared__ __align__(16) i8 sgn[128*64];
    __shared__ int T2l[64];
    int tid = threadIdx.x;
    int rbase = blockIdx.x*4;
    { const v4u* s = (const v4u*)A2;                        v4u* d = (v4u*)a2l;
      for (int i = tid; i < 512; i += 256) d[i] = s[i]; }
    { const v4u* s = (const v4u*)(h1s + (size_t)rbase*1088); v4u* d = (v4u*)slab;
      for (int i = tid; i < 272; i += 256) d[i] = s[i]; }
    if (tid < 64) T2l[tid] = T2[tid];
    __syncthreads();

    int wv = tid >> 5, L = tid & 31;
    int mt = wv & 3, ntb = (wv >> 2)*4;
    int nl = L & 15, hi = L >> 4;
    v8i acc[4] = {};
#pragma unroll
    for (int c = 0; c < 2; c++) {
        v8i af;
        int m = mt*16 + nl;
#pragma unroll
        for (int j = 0; j < 8; j++) {
            int k = (j>>2)*32 + hi*16 + (j&3)*4;
            af[j] = *(const int*)(a2l + m*128 + c*64 + k);
        }
#pragma unroll
        for (int t = 0; t < 4; t++) {
            int col = (ntb + t)*16 + nl;
            int row_l = col >> 5, w = col & 31;
            v8i bf;
#pragma unroll
            for (int j = 0; j < 8; j++) {
                int k = (j>>2)*32 + hi*16 + (j&3)*4;
                int val;
                if (c == 0) { int kw = k>>5, cin = k&31;
                              val = *(const int*)(slab + row_l*1088 + (w+kw)*32 + cin); }
                else        { val = (k < 32) ? *(const int*)(slab + row_l*1088 + (w+2)*32 + k) : 0; }
                bf[j] = val;
            }
            acc[t] = __builtin_amdgcn_wmma_i32_16x16x64_iu8(true, af, true, bf, acc[t], false, false);
        }
    }
#pragma unroll
    for (int t = 0; t < 4; t++) {
        int col = (ntb + t)*16 + nl;
        u32 p0 = 0, p1 = 0;
#pragma unroll
        for (int j = 0; j < 8; j++) {
            int m = mt*16 + hi*8 + j;
            i8 s = (acc[t][j] >= T2l[m]) ? (i8)1 : (i8)-1;
            if (j < 4) p0 |= ((u32)(unsigned char)s) << (8*j);
            else       p1 |= ((u32)(unsigned char)s) << (8*(j-4));
        }
        u32* d = (u32*)(sgn + col*64 + mt*16 + hi*8);
        d[0] = p0; d[1] = p1;
    }
    __syncthreads();
    v4u* ob = (v4u*)(h2s + (size_t)rbase*2176);
    for (int i = tid; i < 544; i += 256) {
        int row_l = i/136, rem = i%136, wp = rem>>2, q = rem&3;
        v4u v = {};
        if (wp >= 1 && wp <= 32) v = ((v4u*)sgn)[(row_l*32 + wp - 1)*4 + q];
        ob[i] = v;
    }
}

__global__ void k_layer3(const i8* __restrict__ h2s, const i8* __restrict__ A3,
                         const int* __restrict__ T3, i8* __restrict__ h3s)
{
    __shared__ __align__(16) i8 a3l[128*192];
    __shared__ __align__(16) i8 slab[4*34*64];
    __shared__ __align__(16) i8 sgn[128*128];
    __shared__ int T3l[128];
    int tid = threadIdx.x;
    int rbase = blockIdx.x*4;
    { const v4u* s = (const v4u*)A3;                         v4u* d = (v4u*)a3l;
      for (int i = tid; i < 1536; i += 256) d[i] = s[i]; }
    { const v4u* s = (const v4u*)(h2s + (size_t)rbase*2176); v4u* d = (v4u*)slab;
      for (int i = tid; i < 544; i += 256) d[i] = s[i]; }
    if (tid < 128) T3l[tid] = T3[tid];
    __syncthreads();

    int wv = tid >> 5, L = tid & 31;
    int mt = wv;
    int nl = L & 15, hi = L >> 4;
    v8i acc[8] = {};
#pragma unroll
    for (int kw = 0; kw < 3; kw++) {
        v8i af;
        int m = mt*16 + nl;
#pragma unroll
        for (int j = 0; j < 8; j++) {
            int k = (j>>2)*32 + hi*16 + (j&3)*4;
            af[j] = *(const int*)(a3l + m*192 + kw*64 + k);
        }
#pragma unroll
        for (int t = 0; t < 8; t++) {
            int col = t*16 + nl;
            int row_l = col >> 5, w = col & 31;
            v8i bf;
#pragma unroll
            for (int j = 0; j < 8; j++) {
                int k = (j>>2)*32 + hi*16 + (j&3)*4;
                bf[j] = *(const int*)(slab + row_l*2176 + (w+kw)*64 + k);
            }
            acc[t] = __builtin_amdgcn_wmma_i32_16x16x64_iu8(true, af, true, bf, acc[t], false, false);
        }
    }
#pragma unroll
    for (int t = 0; t < 8; t++) {
        int col = t*16 + nl;
        u32 p0 = 0, p1 = 0;
#pragma unroll
        for (int j = 0; j < 8; j++) {
            int m = mt*16 + hi*8 + j;
            i8 s = (acc[t][j] >= T3l[m]) ? (i8)1 : (i8)-1;
            if (j < 4) p0 |= ((u32)(unsigned char)s) << (8*j);
            else       p1 |= ((u32)(unsigned char)s) << (8*(j-4));
        }
        u32* d = (u32*)(sgn + col*128 + mt*16 + hi*8);
        d[0] = p0; d[1] = p1;
    }
    __syncthreads();
    v4u* ob = (v4u*)(h3s + (size_t)rbase*2048);
    for (int i = tid; i < 512; i += 256) {
        int row_l = i >> 7, rem = i & 127, wq = rem >> 3, q = rem & 7;
        v4u a = ((v4u*)sgn)[(row_l*32 + wq*2    )*8 + q];
        v4u b = ((v4u*)sgn)[(row_l*32 + wq*2 + 1)*8 + q];
        ob[(row_l*16 + wq)*8 + q] = a & b;
    }
}

__global__ void k_layer4_fc(const i8* __restrict__ h3s, const i8* __restrict__ A4,
                            const float* __restrict__ S4, const float* __restrict__ D4,
                            const float* __restrict__ wfc, const float* __restrict__ bfc,
                            float* __restrict__ out)
{
    __shared__ __align__(16) i8 slab[4*6*16*128];
    __shared__ float s4l[128], d4l[128];
    __shared__ float fcpart[40*4];
    int tid = threadIdx.x;
    int nbase = blockIdx.x*4;
    { const v4u* s = (const v4u*)(h3s + (size_t)nbase*12288); v4u* d = (v4u*)slab;
      for (int i = tid; i < 3072; i += 256) d[i] = s[i]; }
    if (tid < 128) { s4l[tid] = S4[tid]; d4l[tid] = D4[tid]; }
    __syncthreads();

    int wv = tid >> 5, L = tid & 31;
    int mt = wv;
    int nl = L & 15, hi = L >> 4;
    v8i acc[4] = {};
#pragma unroll
    for (int c = 0; c < 12; c++) {
        v8i af;
        int m = mt*16 + nl;
#pragma unroll
        for (int j = 0; j < 8; j++) {
            int k = c*64 + (j>>2)*32 + hi*16 + (j&3)*4;
            af[j] = *(const int*)(A4 + m*768 + k);
        }
#pragma unroll
        for (int t = 0; t < 4; t++) {
            v8i bf;
#pragma unroll
            for (int j = 0; j < 8; j++) {
                int kin = (j>>2)*32 + hi*16 + (j&3)*4;
                int kg = c*64 + kin;
                int h = kg >> 7, cin = kg & 127;
                bf[j] = *(const int*)(slab + t*12288 + h*2048 + nl*128 + cin);
            }
            acc[t] = __builtin_amdgcn_wmma_i32_16x16x64_iu8(true, af, true, bf, acc[t], false, false);
        }
    }
    __syncthreads();
    float* h4 = (float*)slab;
#pragma unroll
    for (int t = 0; t < 4; t++) {
#pragma unroll
        for (int j = 0; j < 8; j++) {
            int m = mt*16 + hi*8 + j;
            float v = (float)acc[t][j]*s4l[m] + d4l[m];
            v = fminf(1.f, fmaxf(-1.f, v));
            h4[t*2048 + m*16 + nl] = v;
        }
    }
    __syncthreads();
    {
        int part = tid >> 6;
        int idx  = tid & 63;
        if (idx < 40) {
            int n_l = idx/10, cls = idx%10;
            float a = 0.f;
            const float* wr = wfc + cls*2048 + part*512;
            const float* hr = h4 + n_l*2048 + part*512;
            for (int q = 0; q < 512; q++) a += hr[q]*wr[q];
            fcpart[idx*4 + part] = a;
        }
    }
    __syncthreads();
    if (tid < 40) {
        int n_l = tid/10, cls = tid%10;
        float a = bfc[cls] + fcpart[tid*4+0] + fcpart[tid*4+1]
                           + fcpart[tid*4+2] + fcpart[tid*4+3];
        out[(size_t)(nbase + n_l)*10 + cls] = a;
    }
}

extern "C" void kernel_launch(void* const* d_in, const int* in_sizes, int n_in,
                              void* d_out, int out_size, void* d_ws, size_t ws_size,
                              hipStream_t stream) {
    const float* x  = (const float*)d_in[0];
    const float* w1 = (const float*)d_in[1];
    const float* w2 = (const float*)d_in[2];
    const float* w3 = (const float*)d_in[3];
    const float* w4 = (const float*)d_in[4];
    const float* b1 = (const float*)d_in[5],  *g1 = (const float*)d_in[6],
               * be1= (const float*)d_in[7],  *m1 = (const float*)d_in[8],  *v1 = (const float*)d_in[9];
    const float* b2 = (const float*)d_in[10], *g2 = (const float*)d_in[11],
               * be2= (const float*)d_in[12], *m2 = (const float*)d_in[13], *v2 = (const float*)d_in[14];
    const float* b3 = (const float*)d_in[15], *g3 = (const float*)d_in[16],
               * be3= (const float*)d_in[17], *m3 = (const float*)d_in[18], *v3 = (const float*)d_in[19];
    const float* b4 = (const float*)d_in[20], *g4 = (const float*)d_in[21],
               * be4= (const float*)d_in[22], *m4 = (const float*)d_in[23], *v4 = (const float*)d_in[24];
    const float* wfc= (const float*)d_in[25];
    const float* bfc= (const float*)d_in[26];

    char* ws = (char*)d_ws;
    i8* A1 = (i8*)(ws + OFF_A1);
    i8* A2 = (i8*)(ws + OFF_A2);
    i8* A3 = (i8*)(ws + OFF_A3);
    i8* A4 = (i8*)(ws + OFF_A4);
    int* T1 = (int*)(ws + OFF_T1);
    int* T2 = (int*)(ws + OFF_T2);
    int* T3 = (int*)(ws + OFF_T3);
    float* S4 = (float*)(ws + OFF_S4);
    float* D4 = (float*)(ws + OFF_D4);
    i8* h1s = (i8*)(ws + OFF_H1S);
    i8* h2s = (i8*)(ws + OFF_H2S);
    i8* h3s = (i8*)(ws + OFF_H3S);

    k_prep_params<<<1, 128, 0, stream>>>(b1,g1,be1,m1,v1, b2,g2,be2,m2,v2,
                                         b3,g3,be3,m3,v3, b4,g4,be4,m4,v4,
                                         T1,T2,T3,S4,D4);
    k_prep_weights<<<384, 256, 0, stream>>>(w1,w2,w3,w4, A1,A2,A3,A4);
    k_layer1<<<6144,  256, 0, stream>>>(x, A1, T1, h1s);
    k_layer2<<<12288, 256, 0, stream>>>(h1s, A2, T2, h2s);
    k_layer3<<<12288, 256, 0, stream>>>(h2s, A3, T3, h3s);
    k_layer4_fc<<<2048, 256, 0, stream>>>(h3s, A4, S4, D4, wfc, bfc, (float*)d_out);
}
